// SMoE_15925738733960
// MI455X (gfx1250) — matrix operand order repaired and checked by reading, not compiled, not run on hardware
//
#include <hip/hip_runtime.h>

// ---------------------------------------------------------------------------
// SMoE fused kernel for gfx1250 (MI455X).  bf16 WMMA (v_wmma_f32_16x16x32_bf16),
// fp32 accumulation.  N=8192 tokens, D=512, DFF=1024, E=8, top-2 gating.
// Rev3: manual double-buffering of B fragments in both GEMM stages so global
// loads for step k+1 overlap the WMMAs of step k (avoids full loadcnt drains
// caused by register reuse).
// ---------------------------------------------------------------------------

typedef float  v8f   __attribute__((ext_vector_type(8)));
typedef __bf16 v16bf __attribute__((ext_vector_type(16)));
typedef __bf16 v8bf  __attribute__((ext_vector_type(8)));
typedef __bf16 v4bf  __attribute__((ext_vector_type(4)));

#define N_TOK 8192
#define DIM   512
#define DFFN  1024
#define NEXP  8

__device__ __forceinline__ __bf16 to_bf16(float f) {
  union { float f; unsigned u; } a; a.f = f;
  unsigned r = a.u + 0x7FFFu + ((a.u >> 16) & 1u);   // round-to-nearest-even
  unsigned short h = (unsigned short)(r >> 16);
  union { unsigned short s; __bf16 b; } o; o.s = h; return o.b;
}

__device__ __forceinline__ v8f wmma_bf16(v16bf a, v16bf b, v8f c) {
  return __builtin_amdgcn_wmma_f32_16x16x32_bf16(
      /*neg_a=*/false, a, /*neg_b=*/false, b,
      /*c_mod=*/(short)0, c, /*reuse_a=*/false, /*reuse_b=*/false);
}

// 16-bit A fragment (16x32): per lane one contiguous 16-element K run.
__device__ __forceinline__ v16bf load_a16(const __bf16* p) {
  v8bf lo = *(const v8bf*)(p);
  v8bf hi = *(const v8bf*)(p + 8);
  v16bf r;
#pragma unroll
  for (int i = 0; i < 8; ++i) { r[i] = lo[i]; r[i + 8] = hi[i]; }
  return r;
}

// ---------------------------------------------------------------------------
// f32 -> bf16 bulk convert (x)
// ---------------------------------------------------------------------------
__global__ __launch_bounds__(256) void k_cvt4(const float4* __restrict__ in,
                                              __bf16* __restrict__ out, int n4) {
  int i = blockIdx.x * 256 + threadIdx.x;
  if (i < n4) {
    float4 v = in[i];
    v4bf o;
    o[0] = to_bf16(v.x); o[1] = to_bf16(v.y);
    o[2] = to_bf16(v.z); o[3] = to_bf16(v.w);
    *(v4bf*)(out + (size_t)i * 4) = o;
  }
}

// ---------------------------------------------------------------------------
// Tiled transpose + convert: in[e][R][C] f32  ->  out[e][C][R] bf16
// (makes the WMMA K dimension contiguous for B fragments)
// ---------------------------------------------------------------------------
__global__ __launch_bounds__(256) void k_transpose(const float* __restrict__ in,
                                                   __bf16* __restrict__ out,
                                                   int R, int C) {
  __shared__ __bf16 tile[32][33];
  const int e = blockIdx.z;
  const float* inp = in + (size_t)e * R * C;
  __bf16* outp = out + (size_t)e * R * C;
  const int c0 = blockIdx.x * 32, r0 = blockIdx.y * 32;
  for (int i = threadIdx.y; i < 32; i += 8)
    tile[i][threadIdx.x] = to_bf16(inp[(size_t)(r0 + i) * C + c0 + threadIdx.x]);
  __syncthreads();
  for (int i = threadIdx.y; i < 32; i += 8)
    outp[(size_t)(c0 + i) * R + r0 + threadIdx.x] = tile[threadIdx.x][i];
}

// ---------------------------------------------------------------------------
// Gating: one wave per token -> top-2 softmax weights per expert  wpe[N][8]
// ---------------------------------------------------------------------------
__global__ __launch_bounds__(256) void k_gate(const float* __restrict__ x,
                                              const float* __restrict__ Wg,
                                              const float* __restrict__ bg,
                                              float* __restrict__ wpe) {
  const int lane = threadIdx.x & 31, wave = threadIdx.x >> 5;
  const int m = blockIdx.x * 8 + wave;
  const float* xr = x + (size_t)m * DIM;
  float acc[NEXP] = {};
  for (int i = 0; i < DIM / 32; ++i) {
    int d = i * 32 + lane;
    float xv = xr[d];
#pragma unroll
    for (int e = 0; e < NEXP; ++e) acc[e] += xv * Wg[d * NEXP + e];
  }
#pragma unroll
  for (int e = 0; e < NEXP; ++e) {
    for (int off = 16; off > 0; off >>= 1)
      acc[e] += __shfl_xor(acc[e], off, 32);
    acc[e] += bg[e];
  }
  int i1 = 0; float v1 = acc[0];
#pragma unroll
  for (int e = 1; e < NEXP; ++e) if (acc[e] > v1) { v1 = acc[e]; i1 = e; }
  int i2 = -1; float v2 = -1e30f;
#pragma unroll
  for (int e = 0; e < NEXP; ++e)
    if (e != i1 && acc[e] > v2) { v2 = acc[e]; i2 = e; }
  float w1 = 1.0f / (1.0f + __expf(v2 - v1));
  float w2 = 1.0f - w1;
  if (lane < NEXP) {
    float w = (lane == i1) ? w1 : (lane == i2) ? w2 : 0.0f;
    wpe[(size_t)m * NEXP + lane] = w;
  }
}

// ---------------------------------------------------------------------------
// Fused MoE: per block = 32 tokens; loop experts; GEMM1 -> silu*mul*gate ->
// GEMM2 accumulate.  8 waves, wave32 WMMA 16x16x32 bf16.  64-wide DFF slices,
// double-buffered B fragments.
// ---------------------------------------------------------------------------
__global__ __launch_bounds__(256) void k_moe(const __bf16* __restrict__ xb,
                                             const __bf16* __restrict__ W1t,
                                             const __bf16* __restrict__ W2t,
                                             const float* __restrict__ b1,
                                             const float* __restrict__ b2,
                                             const float* __restrict__ wpe,
                                             float* __restrict__ out) {
  __shared__ __bf16 XS[32 * 520];        // x tile, padded rows (bank spread)
  __shared__ float  Hh[2 * 32 * 65];     // stage-1 a/g planes [2][32][64]+pad
  __shared__ __bf16 ACT[32 * 72];        // activated tile [32][64]+pad
  __shared__ float  WPE[32 * NEXP];      // gating weights for this token tile
  __shared__ float  B2S[NEXP * DIM];     // b2

  const int tid  = threadIdx.x;
  const int lane = tid & 31;
  const int wave = tid >> 5;
  const int tb   = blockIdx.x * 32;

  // stage x tile rows tb..tb+31 into LDS
  for (int i = tid; i < 32 * 64; i += 256) {
    int r = i >> 6, s = i & 63;
    *(v8bf*)(XS + r * 520 + s * 8) =
        *(const v8bf*)(xb + (size_t)(tb + r) * DIM + s * 8);
  }
  WPE[tid] = wpe[(size_t)tb * NEXP + tid];
  for (int i = tid; i < NEXP * DIM; i += 256) B2S[i] = b2[i];

  v8f Cacc[8];
#pragma unroll
  for (int t = 0; t < 8; ++t) Cacc[t] = (v8f){0, 0, 0, 0, 0, 0, 0, 0};

  const int rg   = wave & 1;             // row group (both stages)
  const int ch   = wave >> 1;            // stage-2 column chunk / stage-1 quad
  const int nlo  = lane & 15;
  const int aoff = (lane >> 4) * 16;     // A fragment K run base
  const int koff = (lane >> 4) * 16;     // B fragment K run base

  __syncthreads();

  for (int e = 0; e < NEXP; ++e) {
    for (int fs = 0; fs < DFFN / 64; ++fs) {
      const int fbase = fs * 64;

      // -------- stage 1: a-tile and g-tile per wave, shared A, K = 512 ----
      const int ca = fbase + ch * 16 + nlo;          // a column
      const float bA = b1[e * (2 * DFFN) + ca];
      const float bG = b1[e * (2 * DFFN) + DFFN + ca];
      v8f accA = (v8f){bA, bA, bA, bA, bA, bA, bA, bA};
      v8f accG = (v8f){bG, bG, bG, bG, bG, bG, bG, bG};
      const __bf16* xrow = XS + (rg * 16 + nlo) * 520;
      const __bf16* wA = W1t + ((size_t)e * (2 * DFFN) + ca) * DIM + koff;
      const __bf16* wG = W1t + ((size_t)e * (2 * DFFN) + DFFN + ca) * DIM + koff;

      // software pipeline: B/A fragments for step ks+1 are loaded into fresh
      // locals before the WMMAs of step ks are issued.
      v16bf Ba = *(const v16bf*)(wA);
      v16bf Bg = *(const v16bf*)(wG);
      v16bf Af = load_a16(xrow + aoff);
#pragma unroll
      for (int ks = 0; ks < DIM / 32; ++ks) {
        v16bf Bac = Ba, Bgc = Bg, Afc = Af;
        if (ks < DIM / 32 - 1) {
          Ba = *(const v16bf*)(wA + (ks + 1) * 32);
          Bg = *(const v16bf*)(wG + (ks + 1) * 32);
          Af = load_a16(xrow + (ks + 1) * 32 + aoff);
        }
        accA = wmma_bf16(Afc, Bac, accA);
        accG = wmma_bf16(Afc, Bgc, accG);
      }
      {
        const int hr0 = rg * 16 + (lane >> 4) * 8;
        const int hc  = ch * 16 + nlo;
#pragma unroll
        for (int v = 0; v < 8; ++v) {
          Hh[(hr0 + v) * 65 + hc]            = accA[v];
          Hh[32 * 65 + (hr0 + v) * 65 + hc]  = accG[v];
        }
      }
      __syncthreads();

      // -------- activation: silu(a)*g, scaled by gating weight ------------
#pragma unroll
      for (int j = 0; j < 8; ++j) {
        int el = j * 256 + tid;                 // 32*64 elements
        int r = el >> 6, c = el & 63;
        float a = Hh[r * 65 + c];
        float g = Hh[32 * 65 + r * 65 + c];
        float s = a / (1.0f + __expf(-a));
        ACT[r * 72 + c] = to_bf16(s * g * WPE[r * NEXP + e]);
      }
      __syncthreads();

      // -------- stage 2: C[32,512] += act[32,64] @ W2[64,512] -------------
      {
        const __bf16* arow = ACT + (rg * 16 + nlo) * 72;
        v16bf A0 = load_a16(arow + aoff);
        v16bf A1 = load_a16(arow + 32 + aoff);
        const __bf16* w2row =
            W2t + ((size_t)e * DIM + (ch * 8) * 16 + nlo) * DFFN + fbase + koff;
        v16bf B0 = *(const v16bf*)(w2row);
        v16bf B1 = *(const v16bf*)(w2row + 32);
#pragma unroll
        for (int t = 0; t < 8; ++t) {
          v16bf B0c = B0, B1c = B1;
          if (t < 7) {
            const __bf16* nrow = w2row + (size_t)(t + 1) * 16 * DFFN;
            B0 = *(const v16bf*)(nrow);
            B1 = *(const v16bf*)(nrow + 32);
          }
          Cacc[t] = wmma_bf16(A0, B0c, Cacc[t]);
          Cacc[t] = wmma_bf16(A1, B1c, Cacc[t]);
        }
      }
      // next iteration's first __syncthreads orders ACT/Hh reuse
    }
  }

  // -------- epilogue: add sum_e w[m,e]*b2[e][d], store fp32 --------
#pragma unroll
  for (int t = 0; t < 8; ++t) {
    const int nt = ch * 8 + t;
#pragma unroll
    for (int v = 0; v < 8; ++v) {
      const int r = rg * 16 + (lane >> 4) * 8 + v;
      const int c = nt * 16 + nlo;
      float bias = 0.0f;
#pragma unroll
      for (int e2 = 0; e2 < NEXP; ++e2)
        bias += WPE[r * NEXP + e2] * B2S[e2 * DIM + c];
      out[(size_t)(tb + r) * DIM + c] = Cacc[t][v] + bias;
    }
  }
}

// ---------------------------------------------------------------------------
extern "C" void kernel_launch(void* const* d_in, const int* in_sizes, int n_in,
                              void* d_out, int out_size, void* d_ws,
                              size_t ws_size, hipStream_t stream) {
  (void)in_sizes; (void)n_in; (void)out_size; (void)ws_size;
  const float* x  = (const float*)d_in[0];
  const float* Wg = (const float*)d_in[1];
  const float* bg = (const float*)d_in[2];
  const float* W1 = (const float*)d_in[3];
  const float* b1 = (const float*)d_in[4];
  const float* W2 = (const float*)d_in[5];
  const float* b2 = (const float*)d_in[6];
  float* out = (float*)d_out;

  char* p = (char*)d_ws;
  __bf16* xb  = (__bf16*)p; p += (size_t)N_TOK * DIM * 2;            // 8 MB
  __bf16* W1t = (__bf16*)p; p += (size_t)NEXP * 2 * DFFN * DIM * 2;  // 16 MB
  __bf16* W2t = (__bf16*)p; p += (size_t)NEXP * DFFN * DIM * 2;      // 8 MB
  float*  wpe = (float*)p;  p += (size_t)N_TOK * NEXP * 4;           // 256 KB

  k_cvt4<<<(N_TOK * DIM / 4 + 255) / 256, 256, 0, stream>>>(
      (const float4*)x, xb, N_TOK * DIM / 4);
  // W1 [E][512][2048] -> W1t [E][2048][512]
  k_transpose<<<dim3(2 * DFFN / 32, DIM / 32, NEXP), dim3(32, 8), 0, stream>>>(
      W1, W1t, DIM, 2 * DFFN);
  // W2 [E][1024][512] -> W2t [E][512][1024]
  k_transpose<<<dim3(DIM / 32, DFFN / 32, NEXP), dim3(32, 8), 0, stream>>>(
      W2, W2t, DFFN, DIM);
  k_gate<<<N_TOK / 8, 256, 0, stream>>>(x, Wg, bg, wpe);
  k_moe<<<N_TOK / 32, 256, 0, stream>>>(xb, W1t, W2t, b1, b2, wpe, out);
}
